// PerceptualHausdorfffLoss_32272384262255
// MI455X (gfx1250) — compile-verified, matrix operand order checked by reading, not run on hardware
//
#include <hip/hip_runtime.h>
#include <hip/hip_bf16.h>
#include <math.h>

// Perceptual Hausdorff loss over one-hot skeleton point sets collapses to a
// set-intersection count per batch:
//   inter_b = sum_t pre_b[t] * gt_b[t]           (masks are exactly 0.0/1.0)
//   loss_b  = (sqrtf(2)-1) * (2*N - 2*inter_b) / N
//   out     = mean_b loss_b
//
// The dot product is reduced through v_wmma_f32_16x16x32_f16 with B == ones:
// every element of D is then a row-sum of A, so the total of all 256 D values
// equals 16 * (sum of all 512 A elements) regardless of the VGPR striping.

#define WH      16384   // 128*128 per image
#define NPTS    2048.0f
#define NBATCH  2

typedef __attribute__((ext_vector_type(16))) _Float16 v16h;
typedef __attribute__((ext_vector_type(8)))  float    v8f;

__global__ __launch_bounds__(64)
void hausdorff_wmma_kernel(const float* __restrict__ pre,
                           const float* __restrict__ gt,
                           float* __restrict__ out)
{
    const int wave = threadIdx.x >> 5;   // 0..1 -> batch index
    const int lane = threadIdx.x & 31;

    const float* __restrict__ p = pre + wave * WH;
    const float* __restrict__ g = gt  + wave * WH;

    // B matrix = all ones (layout-independent).
    v16h ones;
#pragma unroll
    for (int i = 0; i < 16; ++i) ones[i] = (_Float16)1.0f;

    v8f acc = {};

    // 32 WMMA steps x 512 elements = 16384 elements per batch.
    // Lane `lane` supplies 16 contiguous products per step.
#pragma unroll 4
    for (int it = 0; it < WH / 512; ++it) {
        const int base = it * 512 + lane * 16;
        v16h prod;
#pragma unroll
        for (int e = 0; e < 16; e += 4) {
            const float4 pv = *reinterpret_cast<const float4*>(p + base + e);
            const float4 gv = *reinterpret_cast<const float4*>(g + base + e);
            prod[e + 0] = (_Float16)(pv.x * gv.x);
            prod[e + 1] = (_Float16)(pv.y * gv.y);
            prod[e + 2] = (_Float16)(pv.z * gv.z);
            prod[e + 3] = (_Float16)(pv.w * gv.w);
        }
        // acc = prod(16x32) x ones(32x16) + acc   -> every D element is a row sum
        acc = __builtin_amdgcn_wmma_f32_16x16x32_f16(
            /*neg_a=*/false, prod, /*neg_b=*/false, ones,
            /*c_mod=*/(short)0, acc, /*reuse_a=*/false, /*reuse_b=*/false);
    }

    // Per-lane partial total of D.
    float s = acc[0] + acc[1] + acc[2] + acc[3] + acc[4] + acc[5] + acc[6] + acc[7];

    // Wave32 reduction (warpSize == 32 on gfx1250).
#pragma unroll
    for (int off = 16; off >= 1; off >>= 1)
        s += __shfl_xor(s, off, 32);

    __shared__ float inter[NBATCH];
    if (lane == 0) {
        // Each mask row appears in 16 identical D columns -> divide by 16.
        inter[wave] = s * (1.0f / 16.0f);
    }
    __syncthreads();

    if (threadIdx.x == 0) {
        // Match reference fp32 arithmetic: per-miss penalty = sqrtf(2)-1,
        // matches give relu(1e-6 - 1) = 0.
        const float v = sqrtf(2.0f) - 1.0f;
        float loss_sum = 0.0f;
#pragma unroll
        for (int b = 0; b < NBATCH; ++b) {
            const float miss2 = 2.0f * NPTS - 2.0f * inter[b]; // missA + missB
            loss_sum += v * miss2 / NPTS;                      // r1 + r2
        }
        out[0] = loss_sum / (float)NBATCH;                     // mean over batch
    }
}

extern "C" void kernel_launch(void* const* d_in, const int* in_sizes, int n_in,
                              void* d_out, int out_size, void* d_ws, size_t ws_size,
                              hipStream_t stream) {
    (void)in_sizes; (void)n_in; (void)out_size; (void)d_ws; (void)ws_size;
    const float* pre = reinterpret_cast<const float*>(d_in[0]); // [B,1,128,128] fp32
    const float* gt  = reinterpret_cast<const float*>(d_in[1]); // [B,1,128,128] fp32
    float*       out = reinterpret_cast<float*>(d_out);         // scalar fp32

    // One workgroup, 2 waves (one per batch). Problem is latency bound
    // (~256KB total traffic), so a single WGP one-pass kernel is optimal.
    hausdorff_wmma_kernel<<<1, 64, 0, stream>>>(pre, gt, out);
}
